// EquivariantSelfAttention_20031727469179
// MI455X (gfx1250) — matrix operand order repaired and checked by reading, not compiled, not run on hardware
//
#include <hip/hip_runtime.h>
#include <hip/hip_bf16.h>
#include <math.h>

// ---------------- CDNA5 WMMA types ----------------
typedef __attribute__((ext_vector_type(16))) _Float16 v16h;
typedef __attribute__((ext_vector_type(8)))  float    v8f;

// ---------------- constants -----------------------
#define NNODE_FEAT 40          // IN0 + IN1*3 = 16 + 24
#define IN0 16
#define IN1 8
#define NB  10
#define QDIM 20                // K0 + K1*3
#define OUTDIM 40              // O0 + O1*3
#define NTK 20                 // 320 / 16 column tiles (k weights)
#define NTV 40                 // 640 / 16 column tiles (v weights)
#define WAVES 4                // waves per block in edge kernels
#define NCF 92                 // coefficient rows per edge (see layout below)

__device__ __forceinline__ float siluf(float x) {
    // x * rcp(1+e^-x): v_exp_f32 + v_rcp_f32 instead of full-precision divide
    return x * __builtin_amdgcn_rcpf(1.0f + __expf(-x));
}

// order-preserving float<->uint encoding for atomic max
__device__ __forceinline__ unsigned encf(float f) {
    unsigned u = __float_as_uint(f);
    return (u & 0x80000000u) ? ~u : (u | 0x80000000u);
}
__device__ __forceinline__ float decf(unsigned u) {
    return (u & 0x80000000u) ? __uint_as_float(u ^ 0x80000000u)
                             : __uint_as_float(~u);
}

// tensor-product scale factors (m0=16, m1=8 fixed)
#define S000 0.17677669529663687f   // 1/(sqrt(16)*sqrt(2))
#define S110 0.14433756729740643f   // 1/(sqrt(3)*sqrt(8)*sqrt(2))
#define S011 0.14433756729740643f   // 1/(sqrt(16)*sqrt(3))
#define S101 0.20412414523193148f   // 1/(sqrt(8)*sqrt(3))
#define S111 0.14433756729740643f   // 1/(sqrt(2)*sqrt(8)*sqrt(3))
#define INV_S10     0.31622776601683794f
#define INV_SQRT12  0.2886751345948129f
#define INV_SQRT8   0.35355339059327373f

// ---------------- init: zero out / z, set m = -inf (encoded) ------------
__global__ void init_kernel(float* __restrict__ out, float* __restrict__ z,
                            unsigned* __restrict__ menc, int n, int nout) {
    int i = blockIdx.x * blockDim.x + threadIdx.x;
    if (i < nout) out[i] = 0.0f;
    if (i < n) {
        z[i] = 0.0f;
        menc[i] = encf(-3.0e38f);
    }
}

// ---------------- pack Wk2 / Wv2 into 32-lane WMMA B fragments (f16) ----
// frag[(t*32 + lane)*16 + s]:
//   s < 8 : W2[(lane/16)*8 + s][t*16 + lane%16] * 0.25   (K rows 0..15)
//   s >= 8: 0                           (K rows 16..31 of the padded K=32)
// All 32 lanes of the edge kernels load unconditionally -> no exec divergence.
__global__ void pack_kernel(const float* __restrict__ Wk2,
                            const float* __restrict__ Wv2,
                            _Float16* __restrict__ PK,
                            _Float16* __restrict__ PV) {
    int id = blockIdx.x * blockDim.x + threadIdx.x;
    if (id < NTK * 32 * 16) {
        int k = id & 15, ln = (id >> 4) & 31, t = id >> 9;
        PK[id] = (k < 8)
                     ? (_Float16)(Wk2[((ln >> 4) * 8 + k) * (NTK * 16) + t * 16 + (ln & 15)] * 0.25f)
                     : (_Float16)0.0f;
    }
    int id2 = id - NTK * 32 * 16;
    if (id2 >= 0 && id2 < NTV * 32 * 16) {
        int k = id2 & 15, ln = (id2 >> 4) & 31, t = id2 >> 9;
        PV[id2] = (k < 8)
                      ? (_Float16)(Wv2[((ln >> 4) * 8 + k) * (NTV * 16) + t * 16 + (ln & 15)] * 0.25f)
                      : (_Float16)0.0f;
    }
}

// ---------------- q = linear_equiv(feature) ------------------------------
__global__ void q_kernel(const float* __restrict__ feature,
                         const float* __restrict__ Wq0,   // 16 x 8
                         const float* __restrict__ Wq1,   // 8 x 4
                         float* __restrict__ qbuf, int n) {
    int i = blockIdx.x * blockDim.x + threadIdx.x;
    if (i >= n) return;
    const float* x = feature + i * NNODE_FEAT;
    float* q = qbuf + i * QDIM;
#pragma unroll
    for (int o = 0; o < 8; o++) {
        float s = 0.f;
#pragma unroll
        for (int k = 0; k < 16; k++) s += x[k] * Wq0[k * 8 + o];
        q[o] = s * 0.25f;                               // 1/sqrt(16)
    }
#pragma unroll
    for (int o = 0; o < 4; o++)
#pragma unroll
        for (int m = 0; m < 3; m++) {
            float s = 0.f;
#pragma unroll
            for (int k = 0; k < 8; k++) s += x[16 + k * 3 + m] * Wq1[k * 4 + o];
            q[8 + o * 3 + m] = s * INV_SQRT8;
        }
}

// ---------------- edge pass (WMMA FC2 + tensor product) ------------------
// PASS_A: computes k(20), logit, atomicMax(m).   O0=8,  O1=4,  NT=20
// PASS_C: computes v(40), out += alpha*v.        O0=16, O1=8,  NT=40
//
// Coefficient table rows (cfT[cidx][edge], edge-transposed so the 8 rows a
// lane owns are contiguous -> 2x ds_load_b128 per cidx):
//   [ 0..15]  c000[i] = x0[i]*sh0*S000              (region 0)
//   [16..23]  c110[i] = dot(x1[i],sh1)*S110         (region 1)
//   [24..39]  a011[i] = x0[i]*S011                  (region 2, * sh1[m] at flush)
//   [40..63]  x1[i][m]                              (region 3, * sh0*S101 at flush)
//   [64..87]  c111[i][m] = cross(x1[i],sh1)[m]*S111 (region 4)
//   [88..90]  sh1[m]                                (region-2 flush)
//   [91]      sh0*S101                              (region-3 flush)
template <int O0, int O1, int NT, bool PASS_A>
__global__ __launch_bounds__(128)
void edge_pass_kernel(const float* __restrict__ feature,
                      const int*   __restrict__ esrc,
                      const int*   __restrict__ edst,
                      const float* __restrict__ sh,      // E x 4
                      const float* __restrict__ elen,    // E x 10
                      const float* __restrict__ W1,      // 10 x 16
                      const _Float16* __restrict__ Bpack, // NT x 32 x 16 halves
                      const float* __restrict__ qbuf,    // pass A
                      float*       __restrict__ logit,
                      unsigned*    __restrict__ menc,
                      const float* __restrict__ zbuf,    // pass C
                      float*       __restrict__ outbuf)  // pass C
{
    constexpr int NCOMP = O0 + O1 * 3;
    constexpr int B0 = IN0 * O0, B1 = B0 + IN1 * O0;
    constexpr int B2 = B1 + IN0 * O1, B3 = B2 + IN1 * O1;
    constexpr int T0 = B0 / 16, T1 = B1 / 16, T2 = B2 / 16, T3 = B3 / 16;
    static_assert(T0 * 16 == B0 && T1 * 16 == B1 && T2 * 16 == B2 && T3 * 16 == B3,
                  "region boundaries must be tile aligned");

    __shared__ float sW1[NB * 16];
    __shared__ __align__(16) float cfT[WAVES][NCF][16];
    __shared__ __align__(16) float acc[WAVES][16][NCOMP];
    __shared__ float s_alpha[WAVES][16];
    __shared__ int   s_src[WAVES][16];

    const int tid = threadIdx.x;
    const int wave = tid >> 5;
    const int lane = tid & 31;
    const int lane16 = lane & 15;
    const int hi = lane >> 4;                 // 0: rows 0-7 / K 0-7, 1: rows 8-15 / K 8-15
    const int tile = blockIdx.x * WAVES + wave;
    const int ebase = tile * 16;

    // stage FC1 weights
    for (int i = tid; i < NB * 16; i += 128) sW1[i] = W1[i];
    // zero per-edge accumulators
    {
        float* ap = &acc[wave][0][0];
        for (int i = lane; i < 16 * NCOMP; i += 32) ap[i] = 0.0f;
    }
    // per-edge tensor-product coefficients (lanes 0..15, one edge each)
    if (hi == 0) {
        int e = ebase + lane16;
        int d = edst[e];
        const float* x = feature + d * NNODE_FEAT;
        float sh0 = sh[e * 4 + 0];
        float s1x = sh[e * 4 + 1], s1y = sh[e * 4 + 2], s1z = sh[e * 4 + 3];
#define CF(cidx) cfT[wave][cidx][lane16]
#pragma unroll
        for (int i = 0; i < 16; i++) {
            float xi = x[i];
            CF(i)      = xi * sh0 * S000;
            CF(24 + i) = xi * S011;
        }
#pragma unroll
        for (int i = 0; i < 8; i++) {
            float ux = x[16 + i * 3 + 0];
            float uy = x[16 + i * 3 + 1];
            float uz = x[16 + i * 3 + 2];
            CF(16 + i) = (ux * s1x + uy * s1y + uz * s1z) * S110;
            CF(40 + i * 3 + 0) = ux;
            CF(40 + i * 3 + 1) = uy;
            CF(40 + i * 3 + 2) = uz;
            CF(64 + i * 3 + 0) = (uy * s1z - uz * s1y) * S111;  // cross(x1, sh1)
            CF(64 + i * 3 + 1) = (uz * s1x - ux * s1z) * S111;
            CF(64 + i * 3 + 2) = (ux * s1y - uy * s1x) * S111;
        }
        CF(88) = s1x; CF(89) = s1y; CF(90) = s1z;
        CF(91) = sh0 * S101;
#undef CF
        s_src[wave][lane16] = esrc[e];
    }
    __syncthreads();

    // ---- h = silu(elen @ W1 / sqrt(10)) as WMMA A fragment ----
    // lane l<16 : row l, K=0..7 in halves 0..7 ; lane l+16 : row l, K=8..15
    int e = ebase + lane16;
    float el[NB];
#pragma unroll
    for (int b = 0; b < NB; b++) el[b] = elen[e * NB + b];
    v16h a = {};
#pragma unroll
    for (int j = 0; j < 8; j++) {
        int col = hi * 8 + j;
        float d = 0.f;
#pragma unroll
        for (int b = 0; b < NB; b++) d += el[b] * sW1[b * 16 + col];
        a[j] = (_Float16)siluf(d * INV_S10);
    }

    // ---- double-buffered B fragments, unconditional 32-lane loads ----
    auto loadB = [&](int t) -> v16h {
        return *(const v16h*)(Bpack + (t * 32 + lane) * 16);
    };
    v16h bcur = loadB(0);
    __builtin_prefetch(Bpack + (1 * 32 + lane) * 16, 0, 3);
    // issue load for t+1, run WMMA on t, rotate
    auto mma = [&](int t) -> v8f {
        v16h bn = {};
        if (t + 1 < NT) {
            bn = loadB(t + 1);
            if (t + 2 < NT)
                __builtin_prefetch(Bpack + ((t + 2) * 32 + lane) * 16, 0, 3);
        }
        v8f c = {};
        c = __builtin_amdgcn_wmma_f32_16x16x32_f16(
                false, a, false, bcur, (short)0, c, false, false);
        bcur = bn;
        return c;
    };
    // batch-load the 8 row-coefficients at cidx (2x ds_load_b128)
    auto load8 = [&](int cidx, float* co) {
        const float4* p = (const float4*)&cfT[wave][cidx][hi * 8];
        float4 lo = p[0], hi4 = p[1];
        co[0] = lo.x; co[1] = lo.y; co[2] = lo.z; co[3] = lo.w;
        co[4] = hi4.x; co[5] = hi4.y; co[6] = hi4.z; co[7] = hi4.w;
    };

    const int row0 = hi * 8;
    const int ob01 = lane16 & (O0 - 1);       // out0 slot, regions 0+1
    const int o2 = lane16 & (O1 - 1);         // out1 slot, regions 2..4
    float racc[8] = {};                        // regions 0+1 accumulator
    float sacc[8] = {};                        // region 2 scalar accumulator
    float macc[3][8] = {};                     // regions 3 / 4 accumulator

    // region 0: w000  (out0[o] += x0[i]*sh0*w)
#pragma unroll
    for (int t = 0; t < T0; t++) {
        v8f c = mma(t);
        int i = (t * 16 + lane16) / O0;
        float co[8]; load8(i, co);
#pragma unroll
        for (int r = 0; r < 8; r++) racc[r] += co[r] * c[r];
    }
    // region 1: w110  (out0[o] += dot(x1[i],sh1)*w)
#pragma unroll
    for (int t = T0; t < T1; t++) {
        v8f c = mma(t);
        int i = (t * 16 + lane16 - B0) / O0;
        float co[8]; load8(16 + i, co);
#pragma unroll
        for (int r = 0; r < 8; r++) racc[r] += co[r] * c[r];
    }
#pragma unroll
    for (int r = 0; r < 8; r++)
        atomicAdd(&acc[wave][row0 + r][ob01], racc[r]);

    // region 2: w011  (out1[o][m] += x0[i]*sh1[m]*w, separable)
#pragma unroll
    for (int t = T1; t < T2; t++) {
        v8f c = mma(t);
        int i = (t * 16 + lane16 - B1) / O1;
        float co[8]; load8(24 + i, co);
#pragma unroll
        for (int r = 0; r < 8; r++) sacc[r] += co[r] * c[r];
    }
    {
        float shm[3][8];
        load8(88, shm[0]); load8(89, shm[1]); load8(90, shm[2]);
#pragma unroll
        for (int r = 0; r < 8; r++)
#pragma unroll
            for (int m = 0; m < 3; m++)
                atomicAdd(&acc[wave][row0 + r][O0 + o2 * 3 + m], sacc[r] * shm[m][r]);
    }

    // region 3: w101  (out1[o][m] += x1[i][m]*sh0*w, sh0 factored out)
#pragma unroll
    for (int t = T2; t < T3; t++) {
        v8f c = mma(t);
        int i = (t * 16 + lane16 - B2) / O1;
        float co[3][8];
        load8(40 + i * 3 + 0, co[0]);
        load8(40 + i * 3 + 1, co[1]);
        load8(40 + i * 3 + 2, co[2]);
#pragma unroll
        for (int r = 0; r < 8; r++)
#pragma unroll
            for (int m = 0; m < 3; m++) macc[m][r] += co[m][r] * c[r];
    }
    {
        float s0[8]; load8(91, s0);
#pragma unroll
        for (int r = 0; r < 8; r++)
#pragma unroll
            for (int m = 0; m < 3; m++) {
                atomicAdd(&acc[wave][row0 + r][O0 + o2 * 3 + m], macc[m][r] * s0[r]);
                macc[m][r] = 0.0f;             // reuse for region 4
            }
    }

    // region 4: w111  (out1[o][m] += cross(x1[i],sh1)[m]*w)
#pragma unroll
    for (int t = T3; t < NT; t++) {
        v8f c = mma(t);
        int i = (t * 16 + lane16 - B3) / O1;
        float co[3][8];
        load8(64 + i * 3 + 0, co[0]);
        load8(64 + i * 3 + 1, co[1]);
        load8(64 + i * 3 + 2, co[2]);
#pragma unroll
        for (int r = 0; r < 8; r++)
#pragma unroll
            for (int m = 0; m < 3; m++) macc[m][r] += co[m][r] * c[r];
    }
#pragma unroll
    for (int r = 0; r < 8; r++)
#pragma unroll
        for (int m = 0; m < 3; m++)
            atomicAdd(&acc[wave][row0 + r][O0 + o2 * 3 + m], macc[m][r]);

    __syncthreads();

    if constexpr (PASS_A) {
        if (hi == 0) {
            int ee = ebase + lane16;
            int s = s_src[wave][lane16];
            const float* qp = qbuf + s * QDIM;
            const float* kk = acc[wave][lane16];
            float lg = 0.f;
#pragma unroll
            for (int cidx = 0; cidx < NCOMP; cidx++) lg += kk[cidx] * qp[cidx];
            lg *= INV_SQRT12;
            logit[ee] = lg;
            atomicMax(&menc[s], encf(lg));
        }
    } else {
        if (hi == 0) {
            int ee = ebase + lane16;
            int s = s_src[wave][lane16];
            float mv = decf(menc[s]);
            s_alpha[wave][lane16] = __expf(logit[ee] - mv) / zbuf[s];
        }
        __syncthreads();
        for (int idx = lane; idx < 16 * NCOMP; idx += 32) {
            int er = idx / NCOMP, cc = idx - er * NCOMP;
            float v = s_alpha[wave][er] * acc[wave][er][cc];
            atomicAdd(&outbuf[s_src[wave][er] * NCOMP + cc], v);
        }
    }
}

// ---------------- pass B: z[src] += exp(logit - m[src]) ------------------
__global__ void passB_kernel(const int* __restrict__ esrc,
                             const float* __restrict__ logit,
                             const unsigned* __restrict__ menc,
                             float* __restrict__ z, int E) {
    int e = blockIdx.x * blockDim.x + threadIdx.x;
    if (e >= E) return;
    int s = esrc[e];
    atomicAdd(&z[s], __expf(logit[e] - decf(menc[s])));
}

// ---------------- launcher ----------------------------------------------
extern "C" void kernel_launch(void* const* d_in, const int* in_sizes, int n_in,
                              void* d_out, int out_size, void* d_ws, size_t ws_size,
                              hipStream_t stream) {
    const float* feature = (const float*)d_in[0];
    const int*   esrc    = (const int*)  d_in[1];
    const int*   edst    = (const int*)  d_in[2];
    const float* sh      = (const float*)d_in[3];
    const float* elen    = (const float*)d_in[4];
    const float* Wq0     = (const float*)d_in[5];
    const float* Wq1     = (const float*)d_in[6];
    const float* Wk1     = (const float*)d_in[7];
    const float* Wk2     = (const float*)d_in[8];
    const float* Wv1     = (const float*)d_in[9];
    const float* Wv2     = (const float*)d_in[10];
    float* out = (float*)d_out;

    const int E = in_sizes[1];
    const int N = in_sizes[0] / NNODE_FEAT;

    // workspace layout (256-byte aligned sections)
    char* ws = (char*)d_ws;
    size_t off = 0;
    auto alloc = [&](size_t bytes) {
        size_t o = off; off += (bytes + 255) & ~(size_t)255; return o;
    };
    float*    logit = (float*)   (ws + alloc((size_t)E * 4));
    unsigned* menc  = (unsigned*)(ws + alloc((size_t)N * 4));
    float*    zbuf  = (float*)   (ws + alloc((size_t)N * 4));
    float*    qbuf  = (float*)   (ws + alloc((size_t)N * QDIM * 4));
    _Float16* PK    = (_Float16*)(ws + alloc((size_t)NTK * 32 * 16 * 2));
    _Float16* PV    = (_Float16*)(ws + alloc((size_t)NTV * 32 * 16 * 2));
    (void)ws_size; (void)n_in; (void)out_size;

    // 1. init out / z / m
    {
        int items = N * OUTDIM;
        init_kernel<<<(items + 255) / 256, 256, 0, stream>>>(out, zbuf, menc, N, items);
    }
    // 2. pack FC2 weights into 32-lane WMMA B-fragments
    {
        int items = (NTK + NTV) * 32 * 16;
        pack_kernel<<<(items + 255) / 256, 256, 0, stream>>>(Wk2, Wv2, PK, PV);
    }
    // 3. q projection
    q_kernel<<<(N + 255) / 256, 256, 0, stream>>>(feature, Wq0, Wq1, qbuf, N);

    const int tiles = E / 16;                 // E = 256000 -> 16000 tiles
    const int blocks = tiles / WAVES;         // 4000 blocks of 4 waves

    // 4. pass A: k + logit + segment max
    edge_pass_kernel<8, 4, NTK, true><<<blocks, 128, 0, stream>>>(
        feature, esrc, edst, sh, elen, Wk1, PK, qbuf, logit, menc, nullptr, nullptr);

    // 5. pass B: segment sum of exp
    passB_kernel<<<(E + 255) / 256, 256, 0, stream>>>(esrc, logit, menc, zbuf, E);

    // 6. pass C: v + alpha-weighted scatter
    edge_pass_kernel<16, 8, NTV, false><<<blocks, 128, 0, stream>>>(
        feature, esrc, edst, sh, elen, Wv1, PV, nullptr, logit, menc, zbuf, out);
}
